// GCNDeep_15393162789376
// MI455X (gfx1250) — matrix operand order repaired and checked by reading, not compiled, not run on hardware
//
#include <hip/hip_runtime.h>

#define N_NODES 50000
#define N_EDGES 800000
#define NFEAT   128
#define NHID    128
#define NOUT    64

typedef __attribute__((ext_vector_type(16))) _Float16 v16h;
typedef __attribute__((ext_vector_type(8)))  _Float16 v8h;
typedef __attribute__((ext_vector_type(8)))  float    v8f;

// ---------------- fp32 -> fp16 conversion (layer-0 input) -----------------------
__global__ void f32_to_f16_kernel(const float* __restrict__ in,
                                  _Float16* __restrict__ out, long n) {
  long i = (long)blockIdx.x * blockDim.x + threadIdx.x;
  long stride = (long)gridDim.x * blockDim.x;
  for (; i < n; i += stride) out[i] = (_Float16)in[i];
}

// ---------------- pack W (f32, 128 x DOUT) into WMMA B-fragment layout ----------
// Bp[((ct*4 + ks)*32 + lane)*16 + i] = (f16) W[k][col]
//   k = ks*32 + (lane>>4)*16 + i, col = ct*16 + (lane&15)
// so each lane's fragment for (col-tile ct, k-step ks) is 32 contiguous bytes.
__global__ void pack_B_kernel(const float* __restrict__ W,
                              _Float16* __restrict__ Bp, int dout) {
  int idx = blockIdx.x * blockDim.x + threadIdx.x;
  int total = dout * 128;
  if (idx >= total) return;
  int i    = idx & 15;
  int lane = (idx >> 4) & 31;
  int ks   = (idx >> 9) & 3;
  int ct   = idx >> 11;
  int k    = ks * 32 + ((lane >> 4) << 4) + i;
  int col  = ct * 16 + (lane & 15);
  Bp[idx] = (_Float16)W[k * dout + col];
}

// ---------------- WMMA GEMM: C[N x DOUT] = A[N x 128] @ W[128 x DOUT] -----------
// 2 waves/block; each wave computes a 16x32 output slab (two 16x16 tiles sharing
// one A fragment). K=128 -> 4 k-steps -> 8x v_wmma_f32_16x16x32_f16 per wave.
// DOUT is compile-time so all C stores are immediate offsets from one base.
template <int DOUT>
__global__ __launch_bounds__(64) void gemm_f16f32_wmma(
    const _Float16* __restrict__ A, const _Float16* __restrict__ Bp,
    float* __restrict__ C) {
  const int wave = threadIdx.x >> 5;
  const int lane = threadIdx.x & 31;
  const int r    = lane & 15;   // A row / C column within tile
  const int hi   = lane >> 4;   // half-wave selector
  const int row0 = blockIdx.x * 16;
  const int ct0  = blockIdx.y * 4 + wave * 2;  // first of two column tiles

  const _Float16* Arow = A + (size_t)(row0 + r) * NFEAT;
  const _Float16* bp0  = Bp + (((size_t)(ct0 * 4) * 32 + lane) << 4);
  const _Float16* bp1  = Bp + ((((size_t)(ct0 + 1) * 4) * 32 + lane) << 4);
  v8f acc0 = {}, acc1 = {};
#pragma unroll
  for (int ks = 0; ks < 4; ++ks) {
    const int k0 = ks * 32;
    // A fragment: lanes 0-15 hold K={0..15}, lanes 16-31 K={16..31}
    v16h a;
    ((v8h*)&a)[0] = *(const v8h*)(Arow + k0 + 16 * hi);
    ((v8h*)&a)[1] = *(const v8h*)(Arow + k0 + 16 * hi + 8);
    // B fragments: one 32B contiguous chunk per lane from the packed buffer
    v16h b0, b1;
    ((v8h*)&b0)[0] = *(const v8h*)(bp0 + ks * 512);
    ((v8h*)&b0)[1] = *(const v8h*)(bp0 + ks * 512 + 8);
    ((v8h*)&b1)[0] = *(const v8h*)(bp1 + ks * 512);
    ((v8h*)&b1)[1] = *(const v8h*)(bp1 + ks * 512 + 8);
    acc0 = __builtin_amdgcn_wmma_f32_16x16x32_f16(false, a, false, b0,
                                                  (short)0, acc0, false, false);
    acc1 = __builtin_amdgcn_wmma_f32_16x16x32_f16(false, a, false, b1,
                                                  (short)0, acc1, false, false);
  }
  // D layout: VGPR i -> row 8*hi + i, column lane&15
  float* Cp = C + (size_t)(row0 + 8 * hi) * DOUT + ct0 * 16 + r;
#pragma unroll
  for (int i = 0; i < 8; ++i) {
    Cp[i * DOUT]      = acc0[i];
    Cp[i * DOUT + 16] = acc1[i];
  }
}

// ---------------- edge-parallel gather * w -> scatter-add (f32 atomics) ---------
// DOUT compile-time: e/f4 decompose via shift/and; 32 (or 16) consecutive lanes
// cover one edge -> coalesced 512B gathers from L2-resident `support`.
template <int DOUT>
__global__ void edge_scatter_kernel(const float* __restrict__ support,
                                    const int* __restrict__ src,
                                    const int* __restrict__ dst,
                                    const float* __restrict__ w,
                                    float* __restrict__ agg) {
  constexpr int D4 = DOUT / 4;
  const long total = (long)N_EDGES * D4;
  long i = (long)blockIdx.x * blockDim.x + threadIdx.x;
  long stride = (long)gridDim.x * blockDim.x;
  for (; i < total; i += stride) {
    int e  = (int)(i / D4);      // D4 is a power of two -> shift
    int f4 = (int)(i & (D4 - 1));
    int s = src[e], d = dst[e];
    float wt = w[e];
    float4 v = ((const float4*)support)[(size_t)s * D4 + f4];
    float* ap = agg + ((size_t)d * D4 + f4) * 4;
#if defined(__HIP_DEVICE_COMPILE__)
    unsafeAtomicAdd(ap + 0, v.x * wt);
    unsafeAtomicAdd(ap + 1, v.y * wt);
    unsafeAtomicAdd(ap + 2, v.z * wt);
    unsafeAtomicAdd(ap + 3, v.w * wt);
#else
    atomicAdd(ap + 0, v.x * wt);
    atomicAdd(ap + 1, v.y * wt);
    atomicAdd(ap + 2, v.z * wt);
    atomicAdd(ap + 3, v.w * wt);
#endif
  }
}

// ---------------- bias + ReLU fused with f32->f16 for next layer ----------------
template <int DOUT>
__global__ void bias_relu_half_kernel(const float* __restrict__ agg,
                                      const float* __restrict__ bias,
                                      _Float16* __restrict__ out) {
  const long total = (long)N_NODES * DOUT;
  long i = (long)blockIdx.x * blockDim.x + threadIdx.x;
  long stride = (long)gridDim.x * blockDim.x;
  for (; i < total; i += stride) {
    float v = agg[i] + bias[(int)(i & (DOUT - 1))];
    out[i] = (_Float16)fmaxf(v, 0.0f);
  }
}

// ---------------- final layer: bias only, f32 output ----------------------------
template <int DOUT>
__global__ void bias_out_kernel(const float* __restrict__ agg,
                                const float* __restrict__ bias,
                                float* __restrict__ out) {
  const long total = (long)N_NODES * DOUT;
  long i = (long)blockIdx.x * blockDim.x + threadIdx.x;
  long stride = (long)gridDim.x * blockDim.x;
  for (; i < total; i += stride)
    out[i] = agg[i] + bias[(int)(i & (DOUT - 1))];
}

extern "C" void kernel_launch(void* const* d_in, const int* in_sizes, int n_in,
                              void* d_out, int out_size, void* d_ws, size_t ws_size,
                              hipStream_t stream) {
  const float* x    = (const float*)d_in[0];
  const int*   esrc = (const int*)d_in[1];
  const int*   edst = (const int*)d_in[2];
  const float* ew   = (const float*)d_in[3];
  const float* W[4]  = {(const float*)d_in[4], (const float*)d_in[6],
                        (const float*)d_in[8], (const float*)d_in[10]};
  const float* bia[4] = {(const float*)d_in[5], (const float*)d_in[7],
                         (const float*)d_in[9], (const float*)d_in[11]};
  const int din[4]  = {NFEAT, NHID, NHID, NHID};
  const int dOut[4] = {NHID, NHID, NHID, NOUT};

  // ---- workspace layout (all 256B aligned) ----
  char* ws = (char*)d_ws;
  size_t off = 0;
  auto take = [&](size_t bytes) -> char* {
    char* p = ws + off;
    off = (off + bytes + 255) & ~(size_t)255;
    return p;
  };
  _Float16* Bp[4];
  for (int l = 0; l < 4; ++l)
    Bp[l] = (_Float16*)take((size_t)din[l] * dOut[l] * sizeof(_Float16));
  _Float16* h    = (_Float16*)take((size_t)N_NODES * NHID * sizeof(_Float16)); // 12.8 MB
  float* support = (float*)take((size_t)N_NODES * NHID * sizeof(float));       // 25.6 MB
  float* agg     = (float*)take((size_t)N_NODES * NHID * sizeof(float));       // 25.6 MB
  (void)ws_size;

  // ---- pack weights into WMMA B-fragment layout (fused f32->f16) ----
  for (int l = 0; l < 4; ++l) {
    int n = din[l] * dOut[l];
    pack_B_kernel<<<(unsigned)((n + 255) / 256), 256, 0, stream>>>(W[l], Bp[l],
                                                                   dOut[l]);
  }
  // ---- layer-0 input to fp16 ----
  {
    long n = (long)N_NODES * NFEAT;
    f32_to_f16_kernel<<<4096, 256, 0, stream>>>(x, h, n);
  }

  // ---- 4 GCN layers ----
  for (int l = 0; l < 4; ++l) {
    const int dout = dOut[l];

    // support = h @ W  (WMMA f16 in, f32 accum)
    dim3 grid(N_NODES / 16, dout / 64);
    if (dout == NHID)
      gemm_f16f32_wmma<NHID><<<grid, 64, 0, stream>>>(h, Bp[l], support);
    else
      gemm_f16f32_wmma<NOUT><<<grid, 64, 0, stream>>>(h, Bp[l], support);

    // agg = 0
    hipMemsetAsync(agg, 0, (size_t)N_NODES * dout * sizeof(float), stream);

    // agg[dst] += support[src] * w   (edge-parallel, f32 hardware atomics)
    long total = (long)N_EDGES * (dout / 4);
    unsigned blocks = (unsigned)((total + 255) / 256);
    if (dout == NHID)
      edge_scatter_kernel<NHID><<<blocks, 256, 0, stream>>>(support, esrc, edst,
                                                            ew, agg);
    else
      edge_scatter_kernel<NOUT><<<blocks, 256, 0, stream>>>(support, esrc, edst,
                                                            ew, agg);

    if (l < 3) {
      bias_relu_half_kernel<NHID><<<4096, 256, 0, stream>>>(agg, bia[l], h);
    } else {
      bias_out_kernel<NOUT><<<4096, 256, 0, stream>>>(agg, bia[l],
                                                      (float*)d_out);
    }
  }
}
